// GQAAttention_23733989277920
// MI455X (gfx1250) — matrix operand order repaired and checked by reading, not compiled, not run on hardware
//
#include <hip/hip_runtime.h>
#include <hip/hip_bf16.h>

// ---------------------------------------------------------------------------
// GQA attention for gfx1250 (MI455X), bf16 WMMA pipeline, software-pipelined
// with ping-pong fragment buffers and saddr-form addressing.
//   D_MODEL=1024, NUM_HEADS=16, N_GROUP=4, HEAD_DIM=64, KV_DIM=256, BS=2, SEQ=2048
// ---------------------------------------------------------------------------

#define DMODEL 1024
#define KVDIM  256
#define SEQLEN 2048
#define BSZ    2
#define NHEAD  16
#define HDIM   64
#define MTOT   (BSZ * SEQLEN)   // 4096 rows

typedef __attribute__((ext_vector_type(16))) __bf16 v16bf;
typedef __attribute__((ext_vector_type(8)))  __bf16 v8bf;
typedef __attribute__((ext_vector_type(8)))  float  v8f;

__device__ __forceinline__ v8f wmma_bf16(v16bf a, v16bf b, v8f c) {
    // D = A(16x32 bf16) * B(32x16 bf16) + C(16x16 f32)
    return __builtin_amdgcn_wmma_f32_16x16x32_bf16(
        false, a, false, b, (short)0, c, false, false);
}

// A-fragment (16x32, bf16): lane = M, contiguous k order (same as B):
//   e 0..15 -> K = half*16 + e
// base is wave-uniform; off is the per-lane element offset (int -> saddr form).
__device__ __forceinline__ v16bf load_a_frag(const __bf16* base, int off, int half) {
    v8bf lo = *(const v8bf*)(base + off + half * 16);
    v8bf hi = *(const v8bf*)(base + off + 8 + half * 16);
    v16bf f;
#pragma unroll
    for (int i = 0; i < 8; ++i) { f[i] = lo[i]; f[i + 8] = hi[i]; }
    return f;
}

// B-fragment (32x16, bf16): lane = N, K in 16-blocks per lane-half.
__device__ __forceinline__ v16bf load_b_frag(const __bf16* base, int off, int half) {
    return *(const v16bf*)(base + off + half * 16);
}

// ---------------------------------------------------------------------------
// fp32 -> bf16 conversion
// ---------------------------------------------------------------------------
__global__ void cvt_f32_bf16(const float* __restrict__ in, __bf16* __restrict__ out, int n) {
    int i = blockIdx.x * blockDim.x + threadIdx.x;
    if (i < n) out[i] = (__bf16)in[i];
}

// ---------------------------------------------------------------------------
// Projection GEMM:  out = (A[M,1024] @ W[N,1024]^T + bias) * scale   (bf16 out)
// Each wave: 32x64 tile; k-loop unrolled x2 with ping-pong fragment buffers.
// ---------------------------------------------------------------------------
struct Frags { v16bf a0, a1, b[4]; };

__device__ __forceinline__ void load_frags(Frags& f, const __bf16* A, const __bf16* W,
                                           int k, int aoff, int woff, int half) {
    const __bf16* ab = A + k;   // uniform base advances; offsets stay in VGPRs
    const __bf16* wb = W + k;
    f.a0 = load_a_frag(ab, aoff, half);
    f.a1 = load_a_frag(ab, aoff + 16 * DMODEL, half);
#pragma unroll
    for (int t = 0; t < 4; ++t)
        f.b[t] = load_b_frag(wb, woff + t * 16 * DMODEL, half);
}

__device__ __forceinline__ void mma_frags(v8f (&acc)[2][4], const Frags& f) {
#pragma unroll
    for (int t = 0; t < 4; ++t) {
        acc[0][t] = wmma_bf16(f.a0, f.b[t], acc[0][t]);
        acc[1][t] = wmma_bf16(f.a1, f.b[t], acc[1][t]);
    }
}

__global__ void __launch_bounds__(128)
gemm_proj(const __bf16* __restrict__ A, const __bf16* __restrict__ W,
          const float* __restrict__ bias, float scale, int N, int transposedV,
          __bf16* __restrict__ out)
{
    const int lane = threadIdx.x & 31;
    const int wave = threadIdx.x >> 5;
    const int half = lane >> 4;
    const int l15  = lane & 15;
    const int m0 = (blockIdx.x * 4 + wave) * 32;
    const int n0 = blockIdx.y * 64;

    const int aoff = (m0 + l15) * DMODEL;   // per-lane element offsets (fit in i32)
    const int woff = (n0 + l15) * DMODEL;

    v8f acc[2][4] = {};
    Frags f0, f1;

    load_frags(f0, A, W, 0, aoff, woff, half);
    for (int k = 0; k < DMODEL; k += 64) {
        load_frags(f1, A, W, k + 32, aoff, woff, half);
        mma_frags(acc, f0);
        load_frags(f0, A, W, (k + 64) & (DMODEL - 1), aoff, woff, half);  // wraps on last
        mma_frags(acc, f1);
    }

#pragma unroll
    for (int mi = 0; mi < 2; ++mi) {
#pragma unroll
        for (int t = 0; t < 4; ++t) {
            const int n = n0 + t * 16 + l15;       // column held by this lane (D layout)
            const float bv = bias[n];
#pragma unroll
            for (int r = 0; r < 8; ++r) {
                const int m = m0 + mi * 16 + r + 8 * half;
                __bf16 o = (__bf16)((acc[mi][t][r] + bv) * scale);
                if (!transposedV) {
                    out[(size_t)m * N + n] = o;
                } else {
                    const int bb = m >> 11;        // m / SEQLEN
                    const int s  = m & (SEQLEN - 1);
                    out[((size_t)bb * N + n) * SEQLEN + s] = o;
                }
            }
        }
    }
}

// ---------------------------------------------------------------------------
// Flash attention: one wave per 16-row Q tile, KV streamed in tiles of 32.
// kv-loop unrolled x2; K fragments ping-pong, V fragments prefetched behind
// the online softmax.
// ---------------------------------------------------------------------------
__device__ __forceinline__ void load_kfrags(v16bf (&kf)[4], const __bf16* Kb, int kv,
                                            int koff, int half) {
    const __bf16* kn = Kb + kv * KVDIM;   // uniform base
    kf[0] = load_b_frag(kn, koff, half);
    kf[1] = load_b_frag(kn, koff + 32, half);
    kf[2] = load_b_frag(kn, koff + 16 * KVDIM, half);
    kf[3] = load_b_frag(kn, koff + 16 * KVDIM + 32, half);
}

__device__ __forceinline__ void attn_step(
    int kv, int kvn, const v16bf& aq0, const v16bf& aq1,
    v16bf (&kf)[4], v16bf (&kfn)[4], v8f (&acc)[4],
    float (&mi)[8], float (&li)[8],
    const __bf16* Kb, const __bf16* Vb, int koff, int voff,
    __bf16* lp, int half, int l15)
{
    // ---- S = (Q/8) @ K^T over a 16x32 tile ----
    v8f s0 = {}, s1 = {};
    s0 = wmma_bf16(aq0, kf[0], s0);
    s0 = wmma_bf16(aq1, kf[1], s0);
    s1 = wmma_bf16(aq0, kf[2], s1);
    s1 = wmma_bf16(aq1, kf[3], s1);

    // ---- prefetch V (this tile) and K (next tile): hidden behind softmax ----
    v16bf vf[4];
#pragma unroll
    for (int t = 0; t < 4; ++t)
        vf[t] = load_b_frag(Vb + kv, voff + t * 16 * SEQLEN, half);
    load_kfrags(kfn, Kb, kvn, koff, half);

    // ---- online softmax; row m = r + 8*half, cols across the 16-lane half ----
    float mt[8];
#pragma unroll
    for (int r = 0; r < 8; ++r) mt[r] = fmaxf(s0[r], s1[r]);
#pragma unroll
    for (int off = 1; off < 16; off <<= 1) {
#pragma unroll
        for (int r = 0; r < 8; ++r) mt[r] = fmaxf(mt[r], __shfl_xor(mt[r], off, 32));
    }

    float sc[8], rs[8];
#pragma unroll
    for (int r = 0; r < 8; ++r) {
        const float nm = fmaxf(mi[r], mt[r]);
        sc[r] = __expf(mi[r] - nm);
        mi[r] = nm;
        const float p0 = __expf(s0[r] - nm);
        const float p1 = __expf(s1[r] - nm);
        s0[r] = p0; s1[r] = p1;
        rs[r] = p0 + p1;
    }
#pragma unroll
    for (int off = 1; off < 16; off <<= 1) {
#pragma unroll
        for (int r = 0; r < 8; ++r) rs[r] += __shfl_xor(rs[r], off, 32);
    }
#pragma unroll
    for (int r = 0; r < 8; ++r) li[r] = li[r] * sc[r] + rs[r];
#pragma unroll
    for (int t = 0; t < 4; ++t)
#pragma unroll
        for (int r = 0; r < 8; ++r) acc[t][r] *= sc[r];

    // ---- D-layout -> A-layout transpose of P via per-wave LDS ----
#pragma unroll
    for (int r = 0; r < 8; ++r) {
        const int m = r + 8 * half;
        lp[m * 40 + l15]      = (__bf16)s0[r];
        lp[m * 40 + 16 + l15] = (__bf16)s1[r];
    }
    asm volatile("s_wait_dscnt 0" ::: "memory");   // in-order LDS: stores visible
    const v16bf pa = load_a_frag(lp, l15 * 40, half);
    asm volatile("" ::: "memory");

    // ---- O += P @ V ----
#pragma unroll
    for (int t = 0; t < 4; ++t)
        acc[t] = wmma_bf16(pa, vf[t], acc[t]);
}

__global__ void __launch_bounds__(128)
gqa_attn(const __bf16* __restrict__ Q, const __bf16* __restrict__ K,
         const __bf16* __restrict__ Vt, float* __restrict__ out)
{
    __shared__ __bf16 plds[4][16 * 40];        // per-wave 16x32 P tile, stride 40 (80B)

    const int lane = threadIdx.x & 31;
    const int wave = threadIdx.x >> 5;
    const int half = lane >> 4;
    const int l15  = lane & 15;
    const int b  = blockIdx.z;
    const int hh = blockIdx.y;
    const int g  = hh >> 2;                    // KV group (rep = 4)
    const int q0 = (blockIdx.x * 4 + wave) * 16;

    // Q fragments (head_dim 64 = 2 x k32), loaded once.
    const __bf16* qb = Q + (size_t)b * SEQLEN * DMODEL + hh * HDIM;
    const int qoff = (q0 + l15) * DMODEL;
    const v16bf aq0 = load_a_frag(qb, qoff, half);
    const v16bf aq1 = load_a_frag(qb + 32, qoff, half);

    // uniform bases (batch/group folded in) + divergent int offsets
    const __bf16* Kb = K + (size_t)b * SEQLEN * KVDIM + g * HDIM;
    const __bf16* Vb = Vt + ((size_t)b * KVDIM + g * HDIM) * SEQLEN;
    const int koff = l15 * KVDIM;
    const int voff = l15 * SEQLEN;

    v8f acc[4] = {};
    float mi[8], li[8];
#pragma unroll
    for (int r = 0; r < 8; ++r) { mi[r] = -3.0e38f; li[r] = 0.f; }

    __bf16* lp = &plds[wave][0];

    v16bf kf[4], kfn[4];
    load_kfrags(kf, Kb, 0, koff, half);

    for (int kv = 0; kv < SEQLEN; kv += 64) {
        attn_step(kv,      kv + 32,                aq0, aq1, kf,  kfn, acc, mi, li,
                  Kb, Vb, koff, voff, lp, half, l15);
        attn_step(kv + 32, (kv + 64) & (SEQLEN-1), aq0, aq1, kfn, kf,  acc, mi, li,
                  Kb, Vb, koff, voff, lp, half, l15);
    }

    // ---- epilogue: divide by row sum, write fp32 out[b, s, h*64 + d] ----
#pragma unroll
    for (int t = 0; t < 4; ++t) {
#pragma unroll
        for (int r = 0; r < 8; ++r) {
            const int m = q0 + r + 8 * half;
            out[(size_t)(b * SEQLEN + m) * DMODEL + hh * HDIM + t * 16 + l15] =
                acc[t][r] / li[r];
        }
    }
}

// ---------------------------------------------------------------------------
// Host launcher
// ---------------------------------------------------------------------------
extern "C" void kernel_launch(void* const* d_in, const int* in_sizes, int n_in,
                              void* d_out, int out_size, void* d_ws, size_t ws_size,
                              hipStream_t stream) {
    const float* h    = (const float*)d_in[0];
    const float* wq_w = (const float*)d_in[1];
    const float* wq_b = (const float*)d_in[2];
    const float* wk_w = (const float*)d_in[3];
    const float* wk_b = (const float*)d_in[4];
    const float* wv_w = (const float*)d_in[5];
    const float* wv_b = (const float*)d_in[6];
    float* out = (float*)d_out;

    // Workspace carve-up (bf16), ~24.1 MB total.
    __bf16* hbf  = (__bf16*)d_ws;                       // 4096*1024
    __bf16* wqbf = hbf  + (size_t)MTOT * DMODEL;        // 1024*1024
    __bf16* wkbf = wqbf + (size_t)DMODEL * DMODEL;      // 256*1024
    __bf16* wvbf = wkbf + (size_t)KVDIM * DMODEL;       // 256*1024
    __bf16* Qbf  = wvbf + (size_t)KVDIM * DMODEL;       // 4096*1024
    __bf16* Kbf  = Qbf  + (size_t)MTOT * DMODEL;        // 4096*256
    __bf16* Vt   = Kbf  + (size_t)MTOT * KVDIM;         // 2*256*2048

    const int nh = MTOT * DMODEL;
    const int nq = DMODEL * DMODEL;
    const int nk = KVDIM * DMODEL;
    cvt_f32_bf16<<<(nh + 255) / 256, 256, 0, stream>>>(h,    hbf,  nh);
    cvt_f32_bf16<<<(nq + 255) / 256, 256, 0, stream>>>(wq_w, wqbf, nq);
    cvt_f32_bf16<<<(nk + 255) / 256, 256, 0, stream>>>(wk_w, wkbf, nk);
    cvt_f32_bf16<<<(nk + 255) / 256, 256, 0, stream>>>(wv_w, wvbf, nk);

    // Q = (h @ wq^T + b) * 0.125  (softmax scale folded in), [4096,1024] bf16
    gemm_proj<<<dim3(MTOT / 128, DMODEL / 64), 128, 0, stream>>>(
        hbf, wqbf, wq_b, 0.125f, DMODEL, 0, Qbf);
    // K = h @ wk^T + b, [4096,256] bf16
    gemm_proj<<<dim3(MTOT / 128, KVDIM / 64), 128, 0, stream>>>(
        hbf, wkbf, wk_b, 1.0f, KVDIM, 0, Kbf);
    // V^T = (h @ wv^T + b)^T, stored [b, 256, 2048] bf16
    gemm_proj<<<dim3(MTOT / 128, KVDIM / 64), 128, 0, stream>>>(
        hbf, wvbf, wv_b, 1.0f, KVDIM, 1, Vt);

    // Flash attention: 128 q-tiles per (b,h); 4 waves/block, one q-tile per wave.
    gqa_attn<<<dim3(SEQLEN / 64, NHEAD, BSZ), 128, 0, stream>>>(Qbf, Kbf, Vt, out);
}
